// OrganPipe_81346680586805
// MI455X (gfx1250) — matrix operand order repaired and checked by reading, not compiled, not run on hardware
//
#include <hip/hip_runtime.h>
#include <hip/hip_bf16.h>

#define BB 4
#define SS 4096
#define DD 256
#define BSD (BB * SS * DD)

typedef _Float16 half_t;
typedef __attribute__((ext_vector_type(16))) _Float16 v16h;
typedef __attribute__((ext_vector_type(8)))  float    v8f;
typedef int v4i __attribute__((vector_size(16)));   // matches builtin proto

#define AS1 __attribute__((address_space(1)))
#define AS3 __attribute__((address_space(3)))

#if defined(__has_builtin)
#if __has_builtin(__builtin_amdgcn_global_load_async_to_lds_b128)
#define USE_ASYNC 1
#endif
#endif
#ifndef USE_ASYNC
#define USE_ASYNC 0
#endif

// ---------------------------------------------------------------------------
// WMMA helper: D = A(16x32 f16) * B(32x16 f16) + C(16x16 f32)
// ---------------------------------------------------------------------------
__device__ __forceinline__ v8f wmma_f16(v16h a, v16h b, v8f c) {
    return __builtin_amdgcn_wmma_f32_16x16x32_f16(
        /*neg_a=*/false, a, /*neg_b=*/false, b,
        /*c_mod=*/(short)0, c, /*reuse_a=*/false, /*reuse_b=*/false);
}

// ---------------------------------------------------------------------------
// A-fragment loader (16x32, f16). Per ISA 7.12.2: lanes 0-15 row M=lane,
// VGPR v<4 -> K = grp*8 + 2*(v&3)+{0,1}, v>=4 -> K = 16 + grp*8 + 2*(v&3)+{0,1}
// where grp = lane>>4. Consecutive half pairs -> 32-bit loads.
// ---------------------------------------------------------------------------
__device__ __forceinline__ v16h loadA(const half_t* t, int row0, int ld,
                                      int koff, int lane) {
    int row = row0 + (lane & 15);
    int grp = (lane >> 4) & 1;
    union { v16h h; unsigned u[8]; } r;
    const unsigned* p = (const unsigned*)t;
#pragma unroll
    for (int v = 0; v < 8; ++v) {
        int k = koff + ((v < 4) ? 0 : 16) + grp * 8 + 2 * (v & 3);
        r.u[v] = p[(row * ld + k) >> 1];
    }
    return r.h;
}

// ---------------------------------------------------------------------------
// B-fragment loader (32x16, f16): lane = n + 16*kgroup; VGPR v<4 holds
// K = kgroup*8 + 2v,+1; v>=4 holds K = 16 + kgroup*8 + 2(v&3),+1 for column n; [n][k] row-major tile.
// ---------------------------------------------------------------------------
__device__ __forceinline__ v16h loadB(const half_t* t, int n0, int ld,
                                      int koff, int lane) {
    int n  = n0 + (lane & 15);
    int kg = (lane >> 4) & 1;
    union { v16h h; unsigned u[8]; } r;
    const unsigned* p = (const unsigned*)t;
#pragma unroll
    for (int v = 0; v < 8; ++v) {
        int k = koff + ((v < 4) ? 0 : 16) + kg * 8 + 2 * (v & 3);
        r.u[v] = p[(n * ld + k) >> 1];
    }
    return r.h;
}

// ---------------------------------------------------------------------------
// GEMM with bias: out[M,256] = A[M,256] @ W[256,256]^T + b  (einsum bsd,ed->bse)
// ---------------------------------------------------------------------------
template <bool OUT_HALF>
__global__ __launch_bounds__(128) void gemm_bias(const float* __restrict__ A,
                                                 const float* __restrict__ W,
                                                 const float* __restrict__ bias,
                                                 void* __restrict__ out) {
    __shared__ half_t At[64 * 32];
    __shared__ half_t Wt[64 * 32];
    const int m0 = blockIdx.x * 64;
    const int n0 = blockIdx.y * 64;
    const int tid = threadIdx.x, lane = tid & 31, wave = tid >> 5;

    v8f acc[4] = {};
    for (int ks = 0; ks < DD; ks += 32) {
        __syncthreads();
        for (int i = tid; i < 64 * 32; i += 128) {
            int r = i >> 5, c = i & 31;
            At[i] = (half_t)A[(size_t)(m0 + r) * DD + ks + c];
            Wt[i] = (half_t)W[(size_t)(n0 + r) * DD + ks + c];
        }
        if (ks + 32 < DD) {  // prefetch next k-slice into GL2
            int r = tid & 63;
            const float* src = (tid < 64) ? A + (size_t)(m0 + r) * DD
                                          : W + (size_t)(n0 + r) * DD;
            __builtin_prefetch(src + ks + 32, 0, 3);
        }
        __syncthreads();
        v16h af = loadA(At, wave * 16, 32, 0, lane);
        v16h bf[4];
#pragma unroll
        for (int t = 0; t < 4; ++t) bf[t] = loadB(Wt, t * 16, 32, 0, lane);
#pragma unroll
        for (int t = 0; t < 4; ++t) acc[t] = wmma_f16(af, bf[t], acc[t]);
    }
    const int col = lane & 15, g = lane >> 4;
#pragma unroll
    for (int t = 0; t < 4; ++t) {
        float bv = bias[n0 + t * 16 + col];
#pragma unroll
        for (int r = 0; r < 8; ++r) {
            size_t row = (size_t)(m0 + wave * 16 + r + 8 * g);
            size_t cc  = (size_t)(n0 + t * 16 + col);
            float v = acc[t][r] + bv;
            if (OUT_HALF) ((half_t*)out)[row * DD + cc] = (half_t)v;
            else          ((float*) out)[row * DD + cc] = v;
        }
    }
}

// ---------------------------------------------------------------------------
// Flash attention with double-buffered async K tiles.
// Block: 128 threads = 4 waves, 64 queries; wave owns 16 queries x D=256.
// ---------------------------------------------------------------------------
#define KT_BYTES (32 * DD * 2)  // 16 KB per K tile

__global__ __launch_bounds__(128) void flash_attn(const half_t* __restrict__ Qh,
                                                  const half_t* __restrict__ Kh,
                                                  const half_t* __restrict__ Vh,
                                                  float* __restrict__ Out) {
    __shared__ half_t Kt[2][32 * DD];     // [key][d], 2 x 16 KB (double buffer)
    __shared__ half_t Vt[DD * 32];        // [d][key] (transposed), 16 KB
    __shared__ half_t Ps[4][16 * 32];     // per-wave P scratch, 4 KB

    const int b  = blockIdx.x / (SS / 64);
    const int q0 = (blockIdx.x % (SS / 64)) * 64;
    const int tid = threadIdx.x, lane = tid & 31, wave = tid >> 5;
    const size_t base = (size_t)b * SS * DD;
    const int qrow = q0 + wave * 16;

    // K-tile stage-in: async DMA to LDS (ASYNCcnt-tracked), sync fallback.
    auto stage_K = [&](int s, int kb) {
#if USE_ASYNC
        AS1 v4i* src = (AS1 v4i*)(Kh + base + (size_t)kb * DD);
        AS3 v4i* dst = (AS3 v4i*)(&Kt[s][0]);
#pragma unroll
        for (int i = 0; i < 8; ++i) {
            int idx = tid + i * 128;      // 128 threads x 8 x 16B = 16 KB
            __builtin_amdgcn_global_load_async_to_lds_b128(src + idx, dst + idx,
                                                           0, 0);
        }
#else
        const uint4* src = (const uint4*)(Kh + base + (size_t)kb * DD);
        uint4* dst = (uint4*)&Kt[s][0];
        for (int i = tid; i < KT_BYTES / 16; i += 128) dst[i] = src[i];
#endif
    };

    // Q fragments straight from global (f16, row-major)
    v16h qf[8];
#pragma unroll
    for (int ds = 0; ds < 8; ++ds)
        qf[ds] = loadA(Qh + base, qrow, DD, ds * 32, lane);

    float m_run[8], l_run[8];
#pragma unroll
    for (int r = 0; r < 8; ++r) { m_run[r] = -3.0e38f; l_run[r] = 0.0f; }
    v8f O[16] = {};

    stage_K(0, 0);  // prologue: fill buffer 0

    const float scale = 0.0625f;  // 1/sqrt(256)
    for (int kb = 0; kb < SS; kb += 32) {
        const int cur = (kb >> 5) & 1;
        __syncthreads();  // everyone done reading Vt / Kt[cur^1] of prev iter

        // V transpose into LDS (sync; needs VGPR staging anyway)
        for (int i = tid; i < 32 * DD; i += 128) {
            int key = i >> 8, d = i & (DD - 1);
            Vt[d * 32 + key] = Vh[base + (size_t)(kb + key) * DD + d];
        }
        // Kick async stage of the next K tile into the other buffer.
        if (kb + 32 < SS) {
            stage_K(cur ^ 1, kb + 32);
#if USE_ASYNC
            // in-order completion: all but the 8 just-issued are done
            asm volatile("s_wait_asynccnt 0x8" ::: "memory");
#endif
        } else {
#if USE_ASYNC
            asm volatile("s_wait_asynccnt 0x0" ::: "memory");
#endif
        }
        // Prefetch K/V two blocks ahead into GL2.
        if (kb + 64 < SS) {
            const char* pk = (const char*)(Kh + base + (size_t)(kb + 64) * DD);
            const char* pv = (const char*)(Vh + base + (size_t)(kb + 64) * DD);
            __builtin_prefetch(pk + tid * 128, 0, 3);
            __builtin_prefetch(pv + tid * 128, 0, 3);
        }
        __syncthreads();  // Kt[cur] + Vt visible to all waves

        // scores S[16q x 32k]: batch all 8 B-fragment loads, then two
        // independent WMMA accumulation chains (even/odd k-slices).
        v8f Sv[2];
#pragma unroll
        for (int nh = 0; nh < 2; ++nh) {
            v16h bf[8];
#pragma unroll
            for (int ds = 0; ds < 8; ++ds)
                bf[ds] = loadB(Kt[cur], nh * 16, DD, ds * 32, lane);
            v8f s0 = {}, s1 = {};
#pragma unroll
            for (int ds = 0; ds < 8; ds += 2) {
                s0 = wmma_f16(qf[ds], bf[ds], s0);
                s1 = wmma_f16(qf[ds + 1], bf[ds + 1], s1);
            }
            Sv[nh] = s0 + s1;
        }

        // online softmax: row reductions stay inside 16-lane halves
        float corr[8], p0[8], p1[8];
#pragma unroll
        for (int r = 0; r < 8; ++r) {
            float a = Sv[0][r] * scale, c = Sv[1][r] * scale;
            float mx = fmaxf(a, c);
#pragma unroll
            for (int msk = 8; msk >= 1; msk >>= 1)
                mx = fmaxf(mx, __shfl_xor(mx, msk, 32));
            float mn = fmaxf(m_run[r], mx);
            corr[r] = __expf(m_run[r] - mn);
            p0[r] = __expf(a - mn);
            p1[r] = __expf(c - mn);
            float rs = p0[r] + p1[r];
#pragma unroll
            for (int msk = 8; msk >= 1; msk >>= 1)
                rs += __shfl_xor(rs, msk, 32);
            l_run[r] = l_run[r] * corr[r] + rs;
            m_run[r] = mn;
        }
#pragma unroll
        for (int t = 0; t < 16; ++t)
#pragma unroll
            for (int r = 0; r < 8; ++r) O[t][r] *= corr[r];

        // transpose P (C-layout -> A-layout) through per-wave scratch
        const int col = lane & 15, g = lane >> 4;
        half_t* ps = Ps[wave];
#pragma unroll
        for (int r = 0; r < 8; ++r) {
            int row = r + 8 * g;
            ps[row * 32 + col]      = (half_t)p0[r];
            ps[row * 32 + 16 + col] = (half_t)p1[r];
        }
        asm volatile("s_wait_dscnt 0" ::: "memory");  // same-wave DS RAW
        v16h pf = loadA(ps, 0, 32, 0, lane);

        // O += P @ V: 16 independent chains; batch V-fragment loads 4 at a time
#pragma unroll
        for (int tt = 0; tt < 4; ++tt) {
            v16h bf[4];
#pragma unroll
            for (int j = 0; j < 4; ++j)
                bf[j] = loadB(Vt, (tt * 4 + j) * 16, 32, 0, lane);
#pragma unroll
            for (int j = 0; j < 4; ++j)
                O[tt * 4 + j] = wmma_f16(pf, bf[j], O[tt * 4 + j]);
        }
    }

    const int col = lane & 15, g = lane >> 4;
#pragma unroll
    for (int r = 0; r < 8; ++r) {
        float inv = 1.0f / l_run[r];
#pragma unroll
        for (int t = 0; t < 16; ++t)
            Out[base + (size_t)(qrow + r + 8 * g) * DD + t * 16 + col] =
                O[t][r] * inv;
    }
}

// ---------------------------------------------------------------------------
// Gate: p = sigmoid(Wg . mean_s(h2[b]) + bg)
// ---------------------------------------------------------------------------
__global__ __launch_bounds__(256) void gate_kernel(const float* __restrict__ h2,
                                                   const float* __restrict__ Wg,
                                                   const float* __restrict__ bg,
                                                   float* __restrict__ pg) {
    const int b = blockIdx.x, d = threadIdx.x;
    const float* p = h2 + (size_t)b * SS * DD + d;
    float s = 0.0f;
    for (int i = 0; i < SS; ++i) s += p[(size_t)i * DD];
    s *= (1.0f / SS);
    __shared__ float red[DD];
    red[d] = s * Wg[d];
    __syncthreads();
    for (int off = 128; off > 0; off >>= 1) {
        if (d < off) red[d] += red[d + off];
        __syncthreads();
    }
    if (d == 0) pg[b] = 1.0f / (1.0f + __expf(-(red[0] + bg[0])));
}

// ---------------------------------------------------------------------------
// Fused: z = elu(mu + eps*exp(0.5*logvar)); y = x + z; LayerNorm(gamma, beta)
// ---------------------------------------------------------------------------
__global__ __launch_bounds__(256) void reparam_ln(const float* __restrict__ x,
                                                  const float* __restrict__ eps,
                                                  const float* __restrict__ mu,
                                                  const float* __restrict__ lv,
                                                  const float* __restrict__ gamma,
                                                  const float* __restrict__ beta,
                                                  float* __restrict__ out) {
    const size_t row = blockIdx.x;
    const int d = threadIdx.x;
    const size_t idx = row * DD + d;
    float z = mu[idx] + eps[idx] * __expf(0.5f * lv[idx]);
    z = (z > 0.0f) ? z : (__expf(z) - 1.0f);   // ELU(alpha=1)
    float y = x[idx] + z;
    __shared__ float red[DD];
    red[d] = y;
    __syncthreads();
    for (int off = 128; off > 0; off >>= 1) {
        if (d < off) red[d] += red[d + off];
        __syncthreads();
    }
    float mean = red[0] * (1.0f / DD);
    __syncthreads();
    float dy = y - mean;
    red[d] = dy * dy;
    __syncthreads();
    for (int off = 128; off > 0; off >>= 1) {
        if (d < off) red[d] += red[d + off];
        __syncthreads();
    }
    float var = red[0] * (1.0f / DD);
    out[idx] = dy * rsqrtf(var + 1e-5f) * gamma[d] + beta[d];
}

// ---------------------------------------------------------------------------
// Host orchestration
// ---------------------------------------------------------------------------
extern "C" void kernel_launch(void* const* d_in, const int* in_sizes, int n_in,
                              void* d_out, int out_size, void* d_ws, size_t ws_size,
                              hipStream_t stream) {
    const float* x    = (const float*)d_in[0];
    const float* eps  = (const float*)d_in[1];
    const float* Wq1  = (const float*)d_in[2];
    const float* Wk1  = (const float*)d_in[3];
    const float* Wv1  = (const float*)d_in[4];
    const float* Wq2  = (const float*)d_in[5];
    const float* Wk2  = (const float*)d_in[6];
    const float* Wv2  = (const float*)d_in[7];
    const float* Wmu  = (const float*)d_in[8];
    const float* Wlv  = (const float*)d_in[9];
    const float* bq1  = (const float*)d_in[10];
    const float* bk1  = (const float*)d_in[11];
    const float* bv1  = (const float*)d_in[12];
    const float* bq2  = (const float*)d_in[13];
    const float* bk2  = (const float*)d_in[14];
    const float* bv2  = (const float*)d_in[15];
    const float* bmu  = (const float*)d_in[16];
    const float* blv  = (const float*)d_in[17];
    const float* Wg   = (const float*)d_in[18];
    const float* bg   = (const float*)d_in[19];
    const float* gam  = (const float*)d_in[20];
    const float* bet  = (const float*)d_in[21];

    float* out_ln = (float*)d_out;               // [B,S,D]
    float* out_mu = (float*)d_out + (size_t)BSD;
    float* out_lv = (float*)d_out + 2 * (size_t)BSD;
    float* out_pg = (float*)d_out + 3 * (size_t)BSD;

    char* ws = (char*)d_ws;
    half_t* Qh = (half_t*)ws;                     ws += (size_t)BSD * sizeof(half_t);
    half_t* Kh = (half_t*)ws;                     ws += (size_t)BSD * sizeof(half_t);
    half_t* Vh = (half_t*)ws;                     ws += (size_t)BSD * sizeof(half_t);
    float*  h1 = (float*)ws;                      ws += (size_t)BSD * sizeof(float);
    float*  h2 = (float*)ws;                      ws += (size_t)BSD * sizeof(float);

    const dim3 gGemm(BB * SS / 64, DD / 64), bGemm(128);
    const dim3 gAttn(BB * (SS / 64)), bAttn(128);

    // Layer 1
    gemm_bias<true><<<gGemm, bGemm, 0, stream>>>(x, Wq1, bq1, Qh);
    gemm_bias<true><<<gGemm, bGemm, 0, stream>>>(x, Wk1, bk1, Kh);
    gemm_bias<true><<<gGemm, bGemm, 0, stream>>>(x, Wv1, bv1, Vh);
    flash_attn<<<gAttn, bAttn, 0, stream>>>(Qh, Kh, Vh, h1);

    // Layer 2
    gemm_bias<true><<<gGemm, bGemm, 0, stream>>>(h1, Wq2, bq2, Qh);
    gemm_bias<true><<<gGemm, bGemm, 0, stream>>>(h1, Wk2, bk2, Kh);
    gemm_bias<true><<<gGemm, bGemm, 0, stream>>>(h1, Wv2, bv2, Vh);
    flash_attn<<<gAttn, bAttn, 0, stream>>>(Qh, Kh, Vh, h2);

    // Heads + epilogue
    gemm_bias<false><<<gGemm, bGemm, 0, stream>>>(h2, Wmu, bmu, out_mu);
    gemm_bias<false><<<gGemm, bGemm, 0, stream>>>(h2, Wlv, blv, out_lv);
    gate_kernel<<<dim3(BB), dim3(DD), 0, stream>>>(h2, Wg, bg, out_pg);
    reparam_ln<<<dim3(BB * SS), dim3(DD), 0, stream>>>(x, eps, out_mu, out_lv,
                                                       gam, bet, out_ln);
}
